// FinalSeq2Seq_64295660421379
// MI455X (gfx1250) — compile-verified, matrix operand order checked by reading, not run on hardware
//
#include <hip/hip_runtime.h>
#include <math.h>

#define BB 8
#define SS 256
#define TT 128
#define DD 768
#define VV 50257
#define HH 12
#define HDIM 64
#define D3 (3 * DD)
#define RATE_ 1.0f
#define PARAM1_ 0.0f
#define PARAM2_ 1.0f
#define NEGV -1e9f

typedef __bf16 bf16;
typedef __attribute__((ext_vector_type(16))) __bf16 v16bf;
typedef __attribute__((ext_vector_type(8))) float v8f;

// ---------------------------------------------------------------------------
// WMMA helpers (CDNA5 16x16x32 bf16, f32 accumulate)
// ---------------------------------------------------------------------------
__device__ __forceinline__ v8f wmma_bf16(v16bf a, v16bf b, v8f c) {
  return __builtin_amdgcn_wmma_f32_16x16x32_bf16(false, a, false, b, (short)0, c,
                                                 false, false);
}

// Fragment loaders use ROW CLAMPING instead of predication: an out-of-range
// row is clamped to rowMax-1 (loads stay in-bounds); the garbage products only
// reach accumulator elements that the bounded store never writes. This keeps
// the K-loop branch-free (no per-iteration EXEC save/restore or zero fills).
//
// 16-bit A layout (ISA 7.12.2): lane half=lane>>4, row=lane&15;
// elems 0..7 -> K = 8*half..8*half+7 ; elems 8..15 -> K = 16+8*half..+7
__device__ __forceinline__ v16bf frag_rowmajor_f32(const float* __restrict__ base,
                                                   int ld, int row0, int k0,
                                                   int rowMax) {
  int lane = threadIdx.x & 31;
  int half = lane >> 4;
  int row = row0 + (lane & 15);
  row = row < rowMax ? row : rowMax - 1;
  const float* p = base + (long)row * ld + k0 + half * 8;
  v16bf f;
#pragma unroll
  for (int e = 0; e < 8; ++e) f[e] = (bf16)p[e];
#pragma unroll
  for (int e = 0; e < 8; ++e) f[8 + e] = (bf16)p[16 + e];
  return f;
}

__device__ __forceinline__ v16bf frag_rowmajor_bf16(const bf16* __restrict__ base,
                                                    int ld, int row0, int k0,
                                                    int rowMax) {
  int lane = threadIdx.x & 31;
  int half = lane >> 4;
  int row = row0 + (lane & 15);
  row = row < rowMax ? row : rowMax - 1;
  const bf16* p = base + (long)row * ld + k0 + half * 8;
  v16bf f;
#pragma unroll
  for (int e = 0; e < 8; ++e) f[e] = p[e];
#pragma unroll
  for (int e = 0; e < 8; ++e) f[8 + e] = p[16 + e];
  return f;
}

// B fragment where data is stored K-major: element (n,k) at base[k*ld + n].
__device__ __forceinline__ v16bf frag_kmajor_f32(const float* __restrict__ base,
                                                 int ld, int n0, int k0,
                                                 int nMax) {
  int lane = threadIdx.x & 31;
  int half = lane >> 4;
  int n = n0 + (lane & 15);
  n = n < nMax ? n : nMax - 1;
  v16bf f;
#pragma unroll
  for (int e = 0; e < 8; ++e)
    f[e] = (bf16)base[(long)(k0 + half * 8 + e) * ld + n];
#pragma unroll
  for (int e = 0; e < 8; ++e)
    f[8 + e] = (bf16)base[(long)(k0 + 16 + half * 8 + e) * ld + n];
  return f;
}

// C/D layout (ISA 7.12.2): element e -> M = 8*(lane>>4)+e, N = lane&15.
__device__ __forceinline__ void store_tile(v8f acc, float* __restrict__ C,
                                           long ldc, int m0, int M, int n,
                                           int N, const float* __restrict__ bias,
                                           const float* __restrict__ res,
                                           int ldr, float rate) {
  int half = (threadIdx.x & 31) >> 4;
  if (n < N) {
    float bv = bias ? bias[n] : 0.0f;
#pragma unroll
    for (int e = 0; e < 8; ++e) {
      int m = m0 + half * 8 + e;
      if (m < M) {
        float v2 = acc[e] + bv;
        if (res) v2 = res[(long)m * ldr + n] + rate * v2;
        C[(long)m * ldc + n] = v2;
      }
    }
  }
}

// ---------------------------------------------------------------------------
// GEMM (1 tile / wave): C[M,N] = A[M,K](f32) * W[N,K](bf16)^T + bias.
// Used for the tiny sequential GRU-step GEMMs (M=8): maximizes #blocks
// pulling the 3.5 MB L2-resident weight stream -> lowest step latency.
// ---------------------------------------------------------------------------
__global__ void fs_gemm_aw(const float* __restrict__ A, int lda, int M,
                           const bf16* __restrict__ W, int N, int K,
                           const float* __restrict__ bias,
                           const float* __restrict__ res, int ldr, float rate,
                           float* __restrict__ C, long ldc) {
  int wave = threadIdx.x >> 5;
  int n0 = (blockIdx.x * (blockDim.x >> 5) + wave) * 16;
  if (n0 >= N) return;
  int m0 = blockIdx.y * 16;
  v8f acc = {};
  for (int k0 = 0; k0 < K; k0 += 32) {
    v16bf a = frag_rowmajor_f32(A, lda, m0, k0, M);
    v16bf b = frag_rowmajor_bf16(W, K, n0, k0, N);
    acc = wmma_bf16(a, b, acc);
  }
  int n = n0 + ((threadIdx.x & 31) & 15);
  store_tile(acc, C, ldc, m0, M, n, N, bias, res, ldr, rate);
}

// ---------------------------------------------------------------------------
// GEMM (4 N-tiles / wave, 16x64): one A fragment feeds 4 WMMAs -> 2x the
// arithmetic intensity. Used for all batch GEMMs (incl. the two 79-GFLOP
// vocab projections, which are compute-bound: AI ~ M=1024 FLOP/weight-byte).
// ---------------------------------------------------------------------------
__global__ void fs_gemm_aw4(const float* __restrict__ A, int lda, int M,
                            const bf16* __restrict__ W, int N, int K,
                            const float* __restrict__ bias,
                            const float* __restrict__ res, int ldr, float rate,
                            float* __restrict__ C, long ldc) {
  int wave = threadIdx.x >> 5;
  int n0 = (blockIdx.x * (blockDim.x >> 5) + wave) * 64;
  if (n0 >= N) return;
  int m0 = blockIdx.y * 16;
  v8f acc0 = {}, acc1 = {}, acc2 = {}, acc3 = {};
  for (int k0 = 0; k0 < K; k0 += 32) {
    v16bf a = frag_rowmajor_f32(A, lda, m0, k0, M);
    v16bf b0 = frag_rowmajor_bf16(W, K, n0, k0, N);
    v16bf b1 = frag_rowmajor_bf16(W, K, n0 + 16, k0, N);
    v16bf b2 = frag_rowmajor_bf16(W, K, n0 + 32, k0, N);
    v16bf b3 = frag_rowmajor_bf16(W, K, n0 + 48, k0, N);
    acc0 = wmma_bf16(a, b0, acc0);
    acc1 = wmma_bf16(a, b1, acc1);
    acc2 = wmma_bf16(a, b2, acc2);
    acc3 = wmma_bf16(a, b3, acc3);
  }
  int lanecol = (threadIdx.x & 31) & 15;
  store_tile(acc0, C, ldc, m0, M, n0 + lanecol, N, bias, res, ldr, rate);
  store_tile(acc1, C, ldc, m0, M, n0 + 16 + lanecol, N, bias, res, ldr, rate);
  store_tile(acc2, C, ldc, m0, M, n0 + 32 + lanecol, N, bias, res, ldr, rate);
  store_tile(acc3, C, ldc, m0, M, n0 + 48 + lanecol, N, bias, res, ldr, rate);
}

// scores[b,h,s,t] = (q . k) / 8  ; grid: x=1 (8 waves cover T=128), y=S/16, z=B*H
__global__ void fs_gemm_scores(const float* __restrict__ q,
                               const float* __restrict__ k,
                               float* __restrict__ scores) {
  int z = blockIdx.z;
  int b = z / HH;
  int h = z - b * HH;
  int n0 = (threadIdx.x >> 5) * 16;
  int m0 = blockIdx.y * 16;
  const float* qb = q + (long)b * SS * DD + h * HDIM;
  const float* kb = k + (long)b * TT * DD + h * HDIM;
  v8f acc = {};
  for (int k0 = 0; k0 < HDIM; k0 += 32) {
    v16bf a = frag_rowmajor_f32(qb, DD, m0, k0, SS);
    v16bf bb = frag_rowmajor_f32(kb, DD, n0, k0, TT);
    acc = wmma_bf16(a, bb, acc);
  }
  int lane = threadIdx.x & 31;
  int half = lane >> 4;
  int n = n0 + (lane & 15);
  float* cb = scores + (long)(b * HH + h) * SS * TT;
#pragma unroll
  for (int e = 0; e < 8; ++e) {
    int m = m0 + half * 8 + e;
    cb[(long)m * TT + n] = acc[e] * 0.125f;  // 1/sqrt(64)
  }
}

// o[b,s,h,:] = softmax_scores[b,h,s,:] @ v[b,:,h,:] ; grid x=1 (4 waves, HD=64),
// y=S/16, z=B*H ; blockDim 128
__global__ void fs_gemm_av(const float* __restrict__ sm,
                           const float* __restrict__ v,
                           float* __restrict__ o) {
  int z = blockIdx.z;
  int b = z / HH;
  int h = z - b * HH;
  int n0 = (threadIdx.x >> 5) * 16;
  int m0 = blockIdx.y * 16;
  const float* ab = sm + (long)(b * HH + h) * SS * TT;
  const float* vb = v + (long)b * TT * DD + h * HDIM;
  v8f acc = {};
  for (int k0 = 0; k0 < TT; k0 += 32) {
    v16bf a = frag_rowmajor_f32(ab, TT, m0, k0, SS);
    v16bf bb = frag_kmajor_f32(vb, DD, n0, k0, HDIM);
    acc = wmma_bf16(a, bb, acc);
  }
  int lane = threadIdx.x & 31;
  int half = lane >> 4;
  int n = n0 + (lane & 15);
  float* ob = o + (long)b * SS * DD + h * HDIM;
#pragma unroll
  for (int e = 0; e < 8; ++e) {
    int m = m0 + half * 8 + e;
    ob[(long)m * DD + n] = acc[e];
  }
}

// ---------------------------------------------------------------------------
// Elementwise / helper kernels
// ---------------------------------------------------------------------------
__global__ void fs_cvt_bf16(const float* __restrict__ in, bf16* __restrict__ out,
                            long n) {
  long i = (long)blockIdx.x * blockDim.x + threadIdx.x;
  if (i < n) out[i] = (bf16)in[i];
}

__global__ void fs_gather_embed(const int* __restrict__ ids,
                                const float* __restrict__ tab,
                                float* __restrict__ out, long total) {
  long i = (long)blockIdx.x * blockDim.x + threadIdx.x;
  if (i >= total) return;
  long row = i / DD;
  int d = (int)(i - row * DD);
  out[i] = tab[(long)ids[row] * DD + d];
}

__global__ void fs_zero_f32(float* __restrict__ p, long n) {
  long i = (long)blockIdx.x * blockDim.x + threadIdx.x;
  if (i < n) p[i] = 0.0f;
}

__global__ void fs_set_zero_scalar(float* p) {
  if (threadIdx.x == 0 && blockIdx.x == 0) *p = 0.0f;
}

__global__ void fs_combine(const float* __restrict__ a,
                           const float* __restrict__ b, float* __restrict__ c,
                           long n) {
  long i = (long)blockIdx.x * blockDim.x + threadIdx.x;
  if (i < n) c[i] = PARAM1_ * a[i] + PARAM2_ * b[i];
}

__global__ void fs_sqdiff_reduce(const float* __restrict__ a,
                                 const float* __restrict__ b, long n,
                                 float* __restrict__ out) {
  __shared__ float red[256];
  long i = (long)blockIdx.x * blockDim.x + threadIdx.x;
  float v = 0.0f;
  if (i < n) {
    float d = a[i] - b[i];
    v = d * d;
  }
  red[threadIdx.x] = v;
  __syncthreads();
  for (int s = 128; s > 0; s >>= 1) {
    if ((int)threadIdx.x < s) red[threadIdx.x] += red[threadIdx.x + s];
    __syncthreads();
  }
  if (threadIdx.x == 0) atomicAdd(out, red[0]);
}

// GRU cell: mode 0 = no masking, 1 = freeze past len, 2 = freeze + zero output
__global__ void fs_gru_cell(const float* __restrict__ gi, long gi_stride,
                            const float* __restrict__ gh, long gh_stride,
                            const float* __restrict__ hin,
                            float* __restrict__ hout,
                            const int* __restrict__ lens, int t, int mode,
                            float* __restrict__ out, long out_stride) {
  int i = blockIdx.x * blockDim.x + threadIdx.x;
  if (i >= BB * DD) return;
  int b = i / DD;
  int d = i - b * DD;
  const float* gib = gi + (long)b * gi_stride;
  const float* ghb = gh + (long)b * gh_stride;
  float r = 1.0f / (1.0f + expf(-(gib[d] + ghb[d])));
  float z = 1.0f / (1.0f + expf(-(gib[DD + d] + ghb[DD + d])));
  float nn = tanhf(gib[2 * DD + d] + r * ghb[2 * DD + d]);
  float hv = hin[i];
  float hn = (1.0f - z) * nn + z * hv;
  bool valid = (mode == 0) || (t < lens[b]);
  float hnew = valid ? hn : hv;
  hout[i] = hnew;
  if (out) out[(long)b * out_stride + d] = (mode == 2 && !valid) ? 0.0f : hnew;
}

// decode step: oracle = argmax_d(h+g) (softmax/TAU is monotone); x = wte[oracle]
__global__ void fs_decode_select(const float* __restrict__ h,
                                 const float* __restrict__ g,
                                 const float* __restrict__ wte,
                                 float* __restrict__ x) {
  __shared__ float sv[256];
  __shared__ int si[256];
  __shared__ int best;
  int b = blockIdx.x;
  const float* hb = h + (long)b * DD;
  const float* gb = g + (long)b * DD;
  float bv = -INFINITY;
  int bi = 0;
  for (int d = threadIdx.x; d < DD; d += blockDim.x) {
    float v = hb[d] + gb[d];
    if (v > bv) { bv = v; bi = d; }
  }
  sv[threadIdx.x] = bv;
  si[threadIdx.x] = bi;
  __syncthreads();
  for (int s = 128; s > 0; s >>= 1) {
    if ((int)threadIdx.x < s) {
      float ov = sv[threadIdx.x + s];
      int oi = si[threadIdx.x + s];
      if (ov > sv[threadIdx.x] ||
          (ov == sv[threadIdx.x] && oi < si[threadIdx.x])) {
        sv[threadIdx.x] = ov;
        si[threadIdx.x] = oi;
      }
    }
    __syncthreads();
  }
  if (threadIdx.x == 0) best = si[0];
  __syncthreads();
  for (int d = threadIdx.x; d < DD; d += blockDim.x)
    x[(long)b * DD + d] = wte[(long)best * DD + d];
}

// argmax over V for each of B*T rows (first-index tie-break like jnp.argmax)
__global__ void fs_argmax_row(const float* __restrict__ logits, int n,
                              int* __restrict__ out_idx) {
  __shared__ float sv[256];
  __shared__ int si[256];
  int row = blockIdx.x;
  const float* p = logits + (long)row * n;
  float bv = -INFINITY;
  int bi = 0;
  for (int i = threadIdx.x; i < n; i += blockDim.x) {
    float v = p[i];
    if (v > bv) { bv = v; bi = i; }
  }
  sv[threadIdx.x] = bv;
  si[threadIdx.x] = bi;
  __syncthreads();
  for (int s = 128; s > 0; s >>= 1) {
    if ((int)threadIdx.x < s) {
      float ov = sv[threadIdx.x + s];
      int oi = si[threadIdx.x + s];
      if (ov > sv[threadIdx.x] ||
          (ov == sv[threadIdx.x] && oi < si[threadIdx.x])) {
        sv[threadIdx.x] = ov;
        si[threadIdx.x] = oi;
      }
    }
    __syncthreads();
  }
  if (threadIdx.x == 0) out_idx[row] = si[0];
}

// softmax over T=128 with key-padding mask; one block (128 threads) per row
__global__ void fs_attn_softmax(float* __restrict__ scores,
                                const int* __restrict__ mem_lens) {
  __shared__ float red[TT];
  int row = blockIdx.x;  // (b*H + h)*S + s
  int b = row / (HH * SS);
  int t = threadIdx.x;
  float* p = scores + (long)row * TT;
  float v = (t < mem_lens[b]) ? p[t] : NEGV;
  red[t] = v;
  __syncthreads();
  for (int s = TT / 2; s > 0; s >>= 1) {
    if (t < s) red[t] = fmaxf(red[t], red[t + s]);
    __syncthreads();
  }
  float m = red[0];
  __syncthreads();
  float e = expf(v - m);
  red[t] = e;
  __syncthreads();
  for (int s = TT / 2; s > 0; s >>= 1) {
    if (t < s) red[t] += red[t + s];
    __syncthreads();
  }
  p[t] = e / red[0];
}

// ---------------------------------------------------------------------------
// Host orchestration
// ---------------------------------------------------------------------------
extern "C" void kernel_launch(void* const* d_in, const int* in_sizes, int n_in,
                              void* d_out, int out_size, void* d_ws,
                              size_t ws_size, hipStream_t stream) {
  (void)in_sizes; (void)n_in; (void)out_size; (void)ws_size;

  const int* src = (const int*)d_in[0];
  const int* src_len = (const int*)d_in[1];
  const int* dst = (const int*)d_in[2];
  const int* dst_len = (const int*)d_in[3];
  const float* wte = (const float*)d_in[4];
  const float* enc_w_ih = (const float*)d_in[5];
  const float* enc_w_hh = (const float*)d_in[6];
  const float* enc_b_ih = (const float*)d_in[7];
  const float* enc_b_hh = (const float*)d_in[8];
  const float* dec_w_ih = (const float*)d_in[9];
  const float* dec_w_hh = (const float*)d_in[10];
  const float* dec_b_ih = (const float*)d_in[11];
  const float* dec_b_hh = (const float*)d_in[12];
  const float* out_w = (const float*)d_in[13];
  const float* out_b = (const float*)d_in[14];
  const float* ll_w = (const float*)d_in[15];
  const float* ll_b = (const float*)d_in[16];
  const float* attn_in_w = (const float*)d_in[17];
  const float* attn_in_b = (const float*)d_in[18];
  const float* attn_out_w = (const float*)d_in[19];
  const float* attn_out_b = (const float*)d_in[20];
  const float* gumbel = (const float*)d_in[21];

  float* y = (float*)d_out;                      // [B,T,V]; also cand_logits scratch
  float* loss = y + (size_t)BB * TT * VV;        // scalar latent_loss

  // --- workspace bump allocator ---
  char* wp = (char*)d_ws;
  auto alloc = [&](size_t bytes) -> void* {
    void* p = (void*)wp;
    wp += (bytes + 255) & ~(size_t)255;
    return p;
  };

  bf16* outw_b = (bf16*)alloc((size_t)VV * DD * 2);
  bf16* enc_wih_b = (bf16*)alloc((size_t)D3 * DD * 2);
  bf16* enc_whh_b = (bf16*)alloc((size_t)D3 * DD * 2);
  bf16* dec_wih_b = (bf16*)alloc((size_t)D3 * DD * 2);
  bf16* dec_whh_b = (bf16*)alloc((size_t)D3 * DD * 2);
  bf16* attn_inw_b = (bf16*)alloc((size_t)D3 * DD * 2);
  bf16* attn_outw_b = (bf16*)alloc((size_t)DD * DD * 2);
  bf16* llw_b = (bf16*)alloc((size_t)DD * DD * 2);

  float* src_e = (float*)alloc((size_t)BB * SS * DD * 4);
  float* dst_e = (float*)alloc((size_t)BB * TT * DD * 4);
  float* gi_big = (float*)alloc((size_t)BB * SS * D3 * 4);   // enc1 then enc2
  float* gi_dec = (float*)alloc((size_t)BB * TT * D3 * 4);   // dst_e @ dec_w_ih^T
  float* h_cur = (float*)alloc((size_t)BB * DD * 4);
  float* gh_buf = (float*)alloc((size_t)BB * D3 * 4);
  float* gi_buf = (float*)alloc((size_t)BB * D3 * 4);
  float* x_buf = (float*)alloc((size_t)BB * DD * 4);
  float* fout = (float*)alloc((size_t)BB * TT * DD * 4);     // decode hidden states
  int* cand_ids = (int*)alloc((size_t)BB * TT * 4);
  float* cand = (float*)alloc((size_t)BB * TT * DD * 4);
  float* q_buf = (float*)alloc((size_t)BB * SS * DD * 4);
  float* k_buf = (float*)alloc((size_t)BB * TT * DD * 4);
  float* v_buf = (float*)alloc((size_t)BB * TT * DD * 4);
  float* scores = (float*)alloc((size_t)BB * HH * SS * TT * 4);
  float* o_buf = (float*)alloc((size_t)BB * SS * DD * 4);
  float* c_out = (float*)alloc((size_t)BB * SS * DD * 4);
  float* c_aft = (float*)alloc((size_t)BB * SS * DD * 4);
  float* c_mul = (float*)alloc((size_t)BB * SS * DD * 4);
  float* dst_out = (float*)alloc((size_t)BB * TT * DD * 4);

  auto cvt = [&](const float* in, bf16* out, long n) {
    fs_cvt_bf16<<<(unsigned)((n + 255) / 256), 256, 0, stream>>>(in, out, n);
  };
  // small per-step GEMM: 1 tile/wave (max parallelism on the critical path)
  auto gemm1 = [&](const float* A, int lda, int M, const bf16* W, int N, int K,
                   const float* bias, float* C, long ldc) {
    dim3 g((unsigned)((((N + 15) / 16) + 7) / 8), (unsigned)((M + 15) / 16), 1);
    fs_gemm_aw<<<g, 256, 0, stream>>>(A, lda, M, W, N, K, bias, nullptr, 0, 0.f,
                                      C, ldc);
  };
  // batch GEMM: 4 tiles/wave (compute-bound path)
  auto gemm4 = [&](const float* A, int lda, int M, const bf16* W, int N, int K,
                   const float* bias, const float* res, int ldr, float rate,
                   float* C, long ldc) {
    dim3 g((unsigned)((((N + 63) / 64) + 7) / 8), (unsigned)((M + 15) / 16), 1);
    fs_gemm_aw4<<<g, 256, 0, stream>>>(A, lda, M, W, N, K, bias, res, ldr, rate,
                                       C, ldc);
  };
  auto cell = [&](const float* gi, long gis, const float* gh, long ghs,
                  const int* lens, int t, int mode, float* out, long outs) {
    fs_gru_cell<<<(BB * DD + 255) / 256, 256, 0, stream>>>(
        gi, gis, gh, ghs, h_cur, h_cur, lens, t, mode, out, outs);
  };

  // --- 1. weight conversion to bf16 ---
  cvt(out_w, outw_b, (long)VV * DD);
  cvt(enc_w_ih, enc_wih_b, (long)D3 * DD);
  cvt(enc_w_hh, enc_whh_b, (long)D3 * DD);
  cvt(dec_w_ih, dec_wih_b, (long)D3 * DD);
  cvt(dec_w_hh, dec_whh_b, (long)D3 * DD);
  cvt(attn_in_w, attn_inw_b, (long)D3 * DD);
  cvt(attn_out_w, attn_outw_b, (long)DD * DD);
  cvt(ll_w, llw_b, (long)DD * DD);

  // --- 2. embeddings ---
  fs_gather_embed<<<((long)BB * SS * DD + 255) / 256, 256, 0, stream>>>(
      src, wte, src_e, (long)BB * SS * DD);
  fs_gather_embed<<<((long)BB * TT * DD + 255) / 256, 256, 0, stream>>>(
      dst, wte, dst_e, (long)BB * TT * DD);

  // --- 3. encoder pass 1: gi for all t, then sequential gh + cell ---
  gemm4(src_e, DD, BB * SS, enc_wih_b, D3, DD, enc_b_ih, nullptr, 0, 0.f,
        gi_big, D3);
  fs_zero_f32<<<(BB * DD + 255) / 256, 256, 0, stream>>>(h_cur, (long)BB * DD);
  for (int t = 0; t < SS; ++t) {
    gemm1(h_cur, DD, BB, enc_whh_b, D3, DD, enc_b_hh, gh_buf, D3);
    cell(gi_big + (long)t * D3, (long)SS * D3, gh_buf, D3, src_len, t, 1,
         nullptr, 0);
  }

  // --- 4. scheduled-sampling decode ---
  gemm4(dst_e, DD, BB * TT, dec_wih_b, D3, DD, dec_b_ih, nullptr, 0, 0.f,
        gi_dec, D3);
  // h0 = cell(dst_e[:,0], h_enc)
  gemm1(h_cur, DD, BB, dec_whh_b, D3, DD, dec_b_hh, gh_buf, D3);
  cell(gi_dec, (long)TT * D3, gh_buf, D3, nullptr, 0, 0, fout, (long)TT * DD);
  for (int t = 1; t < TT; ++t) {
    fs_decode_select<<<BB, 256, 0, stream>>>(
        h_cur, gumbel + (long)(t - 1) * BB * DD, wte, x_buf);
    gemm1(x_buf, DD, BB, dec_wih_b, D3, DD, dec_b_ih, gi_buf, D3);
    gemm1(h_cur, DD, BB, dec_whh_b, D3, DD, dec_b_hh, gh_buf, D3);
    cell(gi_buf, D3, gh_buf, D3, nullptr, t, 0, fout + (long)t * DD,
         (long)TT * DD);
  }

  // --- 5. candidate logits (into d_out as scratch), argmax, gather ---
  gemm4(fout, DD, BB * TT, outw_b, VV, DD, out_b, nullptr, 0, 0.f, y, VV);
  fs_argmax_row<<<BB * TT, 256, 0, stream>>>(y, VV, cand_ids);
  fs_gather_embed<<<((long)BB * TT * DD + 255) / 256, 256, 0, stream>>>(
      cand_ids, wte, cand, (long)BB * TT * DD);

  // --- 6. decay attention ---
  gemm4(src_e, DD, BB * SS, attn_inw_b, DD, DD, attn_in_b, nullptr, 0, 0.f,
        q_buf, DD);
  gemm4(cand, DD, BB * TT, attn_inw_b + (size_t)DD * DD, DD, DD,
        attn_in_b + DD, nullptr, 0, 0.f, k_buf, DD);
  gemm4(cand, DD, BB * TT, attn_inw_b + (size_t)2 * DD * DD, DD, DD,
        attn_in_b + 2 * DD, nullptr, 0, 0.f, v_buf, DD);
  fs_gemm_scores<<<dim3(1, SS / 16, BB * HH), 256, 0, stream>>>(q_buf, k_buf,
                                                                scores);
  fs_attn_softmax<<<BB * HH * SS, TT, 0, stream>>>(scores, dst_len);
  fs_gemm_av<<<dim3(1, SS / 16, BB * HH), 128, 0, stream>>>(scores, v_buf,
                                                            o_buf);
  gemm4(o_buf, DD, BB * SS, attn_outw_b, DD, DD, attn_out_b, src_e, DD, RATE_,
        c_out, DD);

  // --- 7. c_aft, c_multi, latent loss ---
  gemm4(src_e, DD, BB * SS, llw_b, DD, DD, ll_b, nullptr, 0, 0.f, c_aft, DD);
  fs_combine<<<((long)BB * SS * DD + 255) / 256, 256, 0, stream>>>(
      c_aft, c_out, c_mul, (long)BB * SS * DD);
  fs_set_zero_scalar<<<1, 1, 0, stream>>>(loss);
  fs_sqdiff_reduce<<<((long)BB * SS * DD + 255) / 256, 256, 0, stream>>>(
      c_aft, c_out, (long)BB * SS * DD, loss);

  // --- 8. encoder pass 2 over c_multi ---
  gemm4(c_mul, DD, BB * SS, enc_wih_b, D3, DD, enc_b_ih, nullptr, 0, 0.f,
        gi_big, D3);
  fs_zero_f32<<<(BB * DD + 255) / 256, 256, 0, stream>>>(h_cur, (long)BB * DD);
  for (int t = 0; t < SS; ++t) {
    gemm1(h_cur, DD, BB, enc_whh_b, D3, DD, enc_b_hh, gh_buf, D3);
    cell(gi_big + (long)t * D3, (long)SS * D3, gh_buf, D3, src_len, t, 1,
         nullptr, 0);
  }

  // --- 9. final decoder scan over dst_e from h_enc2 (zero-padded outputs) ---
  for (int t = 0; t < TT; ++t) {
    gemm1(h_cur, DD, BB, dec_whh_b, D3, DD, dec_b_hh, gh_buf, D3);
    cell(gi_dec + (long)t * D3, (long)TT * D3, gh_buf, D3, dst_len, t, 2,
         dst_out + (long)t * DD, (long)TT * DD);
  }

  // --- 10. y = dst_out @ out_w^T + out_b (overwrites cand_logits scratch) ---
  gemm4(dst_out, DD, BB * TT, outw_b, VV, DD, out_b, nullptr, 0, 0.f, y, VV);
}
